// NNCLR_41695542510104
// MI455X (gfx1250) — matrix operand order repaired and checked by reading, not compiled, not run on hardware
//
#include <hip/hip_runtime.h>
#include <hip/hip_bf16.h>
#include <math.h>

// ---------------------------------------------------------------------------
// NNCLR forward loss for MI455X (gfx1250), wave32 + v_wmma_f32_16x16x32_bf16.
// B=512, Q=98304, D=256.  Workspace usage: ~3.03 MB.
// ---------------------------------------------------------------------------

#define BB 512
#define DD 256
#define QQ 98304
#define QBLOCKS 128                 // blocks along queue
#define QTILES_PER_BLOCK ((QQ / 16) / QBLOCKS)   // 48 N-tiles of 16 rows

typedef __attribute__((ext_vector_type(16))) __bf16 v16bf;
typedef __attribute__((ext_vector_type(16))) unsigned short v16us;
typedef __attribute__((ext_vector_type(8)))  float v8f;

union VU { v16us u; v16bf b; };
union VD { unsigned d[8]; v16bf b; };

__device__ __forceinline__ unsigned short f2bf(float f) {
  union { float f; unsigned u; } v; v.f = f;
  unsigned r = v.u + 0x7FFFu + ((v.u >> 16) & 1u);   // round-nearest-even
  return (unsigned short)(r >> 16);
}
__device__ __forceinline__ float bf2f(unsigned short h) {
  union { unsigned u; float f; } v; v.u = ((unsigned)h) << 16;
  return v.f;
}
// pack hi16(f_lo), hi16(f_hi) into one dword with a single v_perm_b32
__device__ __forceinline__ unsigned packbf2(float flo, float fhi) {
  union { float f; unsigned u; } a, b; a.f = flo; b.f = fhi;
  // dst byte0,1 = src1(bytes 2,3) ; dst byte2,3 = src0(bytes 2,3)
  return __builtin_amdgcn_perm(b.u, a.u, 0x07060302u);
}
// monotonic u32 encoding of f32 for unsigned max
__device__ __forceinline__ unsigned fenc(float f) {
  union { float f; unsigned u; } v; v.f = f;
  return (v.u & 0x80000000u) ? ~v.u : (v.u | 0x80000000u);
}

// B operand fragment (32x16 KxN, bf16): lane n<16 holds K=0..15 of column n,
// lane n+16 holds K=16..31 -> 16 consecutive floats.  Truncating conversion:
// 4x b128 load + 8x v_perm_b32, no other VALU.
__device__ __forceinline__ v16bf loadB_trunc(const float* __restrict__ p) {
  const float4 f0 = ((const float4*)p)[0];
  const float4 f1 = ((const float4*)p)[1];
  const float4 f2 = ((const float4*)p)[2];
  const float4 f3 = ((const float4*)p)[3];
  VD v;
  v.d[0] = packbf2(f0.x, f0.y); v.d[1] = packbf2(f0.z, f0.w);
  v.d[2] = packbf2(f1.x, f1.y); v.d[3] = packbf2(f1.z, f1.w);
  v.d[4] = packbf2(f2.x, f2.y); v.d[5] = packbf2(f2.z, f2.w);
  v.d[6] = packbf2(f3.x, f3.y); v.d[7] = packbf2(f3.z, f3.w);
  return v.b;
}

// A operand fragment (16x32 MxK, bf16): lane m (h=0) holds K 0..15,
// lane m+16 (h=1) holds K 16..31.  RNE (one-time, cold path).
__device__ __forceinline__ v16bf loadA_bf(const float* __restrict__ row, int k0, int h) {
  VU v;
  const float* lo = row + k0 + 16 * h;
  const float* hi = row + k0 + 16 * h + 8;
  float4 a = *(const float4*)(lo),     b = *(const float4*)(lo + 4);
  float4 c = *(const float4*)(hi),     d = *(const float4*)(hi + 4);
  v.u[0] = f2bf(a.x); v.u[1] = f2bf(a.y); v.u[2]  = f2bf(a.z); v.u[3]  = f2bf(a.w);
  v.u[4] = f2bf(b.x); v.u[5] = f2bf(b.y); v.u[6]  = f2bf(b.z); v.u[7]  = f2bf(b.w);
  v.u[8] = f2bf(c.x); v.u[9] = f2bf(c.y); v.u[10] = f2bf(c.z); v.u[11] = f2bf(c.w);
  v.u[12]= f2bf(d.x); v.u[13]= f2bf(d.y); v.u[14] = f2bf(d.z); v.u[15] = f2bf(d.w);
  return v.b;
}

// hi/lo split versions (for ~f32-accurate small GEMMs)
struct Frag2 { v16bf hi; v16bf lo; };
__device__ __forceinline__ void split1(float f, unsigned short& h, unsigned short& l) {
  h = f2bf(f);
  l = f2bf(f - bf2f(h));
}
__device__ __forceinline__ Frag2 load16bf2(const float* __restrict__ p) {
  VU vh, vl;
#pragma unroll
  for (int i = 0; i < 16; i += 4) {
    float4 f = *(const float4*)(p + i);
    unsigned short h, l;
    split1(f.x, h, l); vh.u[i + 0] = h; vl.u[i + 0] = l;
    split1(f.y, h, l); vh.u[i + 1] = h; vl.u[i + 1] = l;
    split1(f.z, h, l); vh.u[i + 2] = h; vl.u[i + 2] = l;
    split1(f.w, h, l); vh.u[i + 3] = h; vl.u[i + 3] = l;
  }
  Frag2 r; r.hi = vh.b; r.lo = vl.b; return r;
}
__device__ __forceinline__ Frag2 loadA_bf2(const float* __restrict__ row, int k0, int h) {
  VU vh, vl;
  const float* lo = row + k0 + 16 * h;
  const float* hi = row + k0 + 16 * h + 8;
  float buf[16];
  *(float4*)(buf + 0)  = *(const float4*)(lo);
  *(float4*)(buf + 4)  = *(const float4*)(lo + 4);
  *(float4*)(buf + 8)  = *(const float4*)(hi);
  *(float4*)(buf + 12) = *(const float4*)(hi + 4);
#pragma unroll
  for (int i = 0; i < 16; ++i) {
    unsigned short a, b;
    split1(buf[i], a, b);
    vh.u[i] = a; vl.u[i] = b;
  }
  Frag2 r; r.hi = vh.b; r.lo = vl.b; return r;
}

// ---------------------------------------------------------------------------
// K1: row L2-normalize (f32) + zero the packed argmax table.
// grid 1024 blocks x 256 threads; blocks 0..511 -> p1, 512..1023 -> p2.
// ---------------------------------------------------------------------------
__global__ __launch_bounds__(256) void nnclr_normalize(
    const float* __restrict__ p1, const float* __restrict__ p2,
    float* __restrict__ o1, float* __restrict__ o2,
    unsigned long long* __restrict__ best) {
  const int t = threadIdx.x;
  const int blk = blockIdx.x;
  if (blk < 4) best[blk * 256 + t] = 0ull;   // 1024 packed (val,idx) slots

  const float* src = (blk < BB) ? p1 : p2;
  float* dst       = (blk < BB) ? o1 : o2;
  const int row = blk & (BB - 1);
  const float x = src[(size_t)row * DD + t];
  __shared__ float sh[256];
  sh[t] = x * x;
  __syncthreads();
#pragma unroll
  for (int s = 128; s > 0; s >>= 1) {
    if (t < s) sh[t] += sh[t + s];
    __syncthreads();
  }
  const float den = fmaxf(sqrtf(sh[0]), 1e-12f);
  dst[(size_t)row * DD + t] = x / den;
}

// ---------------------------------------------------------------------------
// K2: fused sims + argmax for both projections.
// grid (QBLOCKS, 2) x 512 threads (16 waves).  wave w owns p-row-tile
// blockIdx.y*16+w for BOTH p1 and p2; streams 48 queue N-tiles; K=256 as
// 8 x v_wmma_f32_16x16x32_bf16 per tile per projection.  Hot loop per K-step:
// 4x global_load_b128 + 8x v_perm_b32 + 2x v_wmma.
// ---------------------------------------------------------------------------
__global__ __launch_bounds__(512) void nnclr_argmax(
    const float* __restrict__ queue, const float* __restrict__ p1n,
    const float* __restrict__ p2n, unsigned long long* __restrict__ best) {
  const int lane = threadIdx.x & 31;
  const int wave = threadIdx.x >> 5;         // 0..15
  const int n = lane & 15;
  const int h = lane >> 4;
  const int mtile = blockIdx.y * 16 + wave;  // 0..31

  const float* r1 = p1n + (size_t)(mtile * 16 + n) * DD;
  const float* r2 = p2n + (size_t)(mtile * 16 + n) * DD;
  v16bf a1[8], a2[8];
#pragma unroll
  for (int ks = 0; ks < 8; ++ks) {
    a1[ks] = loadA_bf(r1, ks * 32, h);
    a2[ks] = loadA_bf(r2, ks * 32, h);
  }

  float bv1[8], bv2[8];
  unsigned bi1[8], bi2[8];
#pragma unroll
  for (int r = 0; r < 8; ++r) { bv1[r] = -INFINITY; bv2[r] = -INFINITY; bi1[r] = 0u; bi2[r] = 0u; }

  const int t0 = blockIdx.x * QTILES_PER_BLOCK;
  for (int t = 0; t < QTILES_PER_BLOCK; ++t) {
    const int q0 = (t0 + t) * 16;
    const float* brow = queue + (size_t)(q0 + n) * DD + 16 * h;
    if (t + 1 < QTILES_PER_BLOCK) {
      // global_prefetch_b8 of next tile's row (speculative, cacheline pull)
      __builtin_prefetch(brow + 16 * DD, 0, 0);
    }
    v8f acc1 = {}; v8f acc2 = {};
#pragma unroll
    for (int ks = 0; ks < 8; ++ks) {
      v16bf bm = loadB_trunc(brow + ks * 32);
      acc1 = __builtin_amdgcn_wmma_f32_16x16x32_bf16(false, a1[ks], false, bm,
                                                     (short)0, acc1, false, false);
      acc2 = __builtin_amdgcn_wmma_f32_16x16x32_bf16(false, a2[ks], false, bm,
                                                     (short)0, acc2, false, false);
    }
    const unsigned q = (unsigned)(q0 + n);   // this lane's candidate column
#pragma unroll
    for (int r = 0; r < 8; ++r) {            // q increases monotonically -> '>' keeps first max
      float v1 = acc1[r]; if (v1 > bv1[r]) { bv1[r] = v1; bi1[r] = q; }
      float v2 = acc2[r]; if (v2 > bv2[r]) { bv2[r] = v2; bi2[r] = q; }
    }
  }

  // reduce across the 16 lanes of each N-group, then global packed atomicMax
#pragma unroll
  for (int r = 0; r < 8; ++r) {
    float v1 = bv1[r]; unsigned i1 = bi1[r];
    float v2 = bv2[r]; unsigned i2 = bi2[r];
#pragma unroll
    for (int m = 8; m >= 1; m >>= 1) {
      float ov = __shfl_xor(v1, m); unsigned oi = (unsigned)__shfl_xor((int)i1, m);
      if (ov > v1 || (ov == v1 && oi < i1)) { v1 = ov; i1 = oi; }
      ov = __shfl_xor(v2, m); oi = (unsigned)__shfl_xor((int)i2, m);
      if (ov > v2 || (ov == v2 && oi < i2)) { v2 = ov; i2 = oi; }
    }
    if (n == 0) {
      const int row = mtile * 16 + r + 8 * h;
      const unsigned long long k1 = ((unsigned long long)fenc(v1) << 32) | (unsigned)(~i1);
      const unsigned long long k2 = ((unsigned long long)fenc(v2) << 32) | (unsigned)(~i2);
      atomicMax(&best[row], k1);
      atomicMax(&best[BB + row], k2);
    }
  }
}

// ---------------------------------------------------------------------------
// K3: M1 = nn1 @ p2n.T, M2 = nn2 @ p1n.T, scaled by 1/T, hi/lo bf16 split
// (3 WMMAs per K-step) for ~f32 precision.  grid (32, 4, 2) x 256 threads.
// ---------------------------------------------------------------------------
__global__ __launch_bounds__(256) void nnclr_logits(
    const float* __restrict__ queue, const float* __restrict__ p1n,
    const float* __restrict__ p2n, const unsigned long long* __restrict__ best,
    float* __restrict__ Mbuf, const float* __restrict__ tptr) {
  const int lane = threadIdx.x & 31;
  const int wave = threadIdx.x >> 5;               // 0..7
  const int n = lane & 15;
  const int h = lane >> 4;
  const int rowTile = blockIdx.x;                  // 0..31
  const int colTile = blockIdx.y * 8 + wave;       // 0..31
  const int mat = blockIdx.z;                      // 0: (nn1,p2)  1: (nn2,p1)

  const int arow = rowTile * 16 + n;
  const unsigned idx = ~(unsigned)(best[mat * BB + arow] & 0xffffffffull);
  const float* ar = queue + (size_t)idx * DD;
  const float* br = (mat ? p1n : p2n) + (size_t)(colTile * 16 + n) * DD + 16 * h;

  v8f acc = {};
#pragma unroll
  for (int ks = 0; ks < 8; ++ks) {
    Frag2 am = loadA_bf2(ar, ks * 32, h);
    Frag2 bm = load16bf2(br + ks * 32);
    acc = __builtin_amdgcn_wmma_f32_16x16x32_bf16(false, am.hi, false, bm.hi,
                                                  (short)0, acc, false, false);
    acc = __builtin_amdgcn_wmma_f32_16x16x32_bf16(false, am.hi, false, bm.lo,
                                                  (short)0, acc, false, false);
    acc = __builtin_amdgcn_wmma_f32_16x16x32_bf16(false, am.lo, false, bm.hi,
                                                  (short)0, acc, false, false);
  }
  const float invT = 1.0f / tptr[0];
  float* Mp = Mbuf + (size_t)mat * BB * BB;
#pragma unroll
  for (int r = 0; r < 8; ++r) {
    const int m = rowTile * 16 + r + 8 * h;
    Mp[(size_t)m * BB + colTile * 16 + n] = acc[r] * invT;
  }
}

// ---------------------------------------------------------------------------
// K4: loss[t] = logsumexp(row/col of M) - diag.  grid 2048 x 256 threads.
// seg 0: M1 row, 1: M1 col, 2: M2 row, 3: M2 col (matches concat order).
// ---------------------------------------------------------------------------
__global__ __launch_bounds__(256) void nnclr_loss(
    const float* __restrict__ Mbuf, float* __restrict__ out) {
  const int t = blockIdx.x;
  const int seg = t >> 9;
  const int i = t & (BB - 1);
  const int mat = seg >> 1;
  const int isCol = seg & 1;
  const float* Mm = Mbuf + (size_t)mat * BB * BB;
  const int tid = threadIdx.x;

  const int j0 = tid, j1 = tid + 256;
  const float v0 = isCol ? Mm[(size_t)j0 * BB + i] : Mm[(size_t)i * BB + j0];
  const float v1 = isCol ? Mm[(size_t)j1 * BB + i] : Mm[(size_t)i * BB + j1];

  __shared__ float sh[256];
  sh[tid] = fmaxf(v0, v1);
  __syncthreads();
#pragma unroll
  for (int s = 128; s > 0; s >>= 1) {
    if (tid < s) sh[tid] = fmaxf(sh[tid], sh[tid + s]);
    __syncthreads();
  }
  const float mx = sh[0];
  __syncthreads();
  sh[tid] = expf(v0 - mx) + expf(v1 - mx);
  __syncthreads();
#pragma unroll
  for (int s = 128; s > 0; s >>= 1) {
    if (tid < s) sh[tid] += sh[tid + s];
    __syncthreads();
  }
  if (tid == 0) {
    const float diag = Mm[(size_t)i * BB + i];
    out[t] = logf(sh[0]) + mx - diag;
  }
}

// ---------------------------------------------------------------------------
extern "C" void kernel_launch(void* const* d_in, const int* in_sizes, int n_in,
                              void* d_out, int out_size, void* d_ws, size_t ws_size,
                              hipStream_t stream) {
  (void)in_sizes; (void)n_in; (void)out_size; (void)ws_size;
  const float* p1    = (const float*)d_in[0];   // [512,256]
  const float* p2    = (const float*)d_in[1];   // [512,256]
  const float* queue = (const float*)d_in[2];   // [98304,256] (pre-normalized)
  const float* temp  = (const float*)d_in[3];   // scalar
  float* out = (float*)d_out;                   // [2048]

  // workspace layout (~3.03 MB)
  float* p1n = (float*)d_ws;                                   // 512*256 f32
  float* p2n = p1n + BB * DD;                                  // 512*256 f32
  unsigned long long* best = (unsigned long long*)(p2n + BB * DD); // 1024 u64
  float* Mbuf = (float*)(best + 2 * BB);                       // 2*512*512 f32

  nnclr_normalize<<<2 * BB, 256, 0, stream>>>(p1, p2, p1n, p2n, best);
  nnclr_argmax<<<dim3(QBLOCKS, 2, 1), 512, 0, stream>>>(queue, p1n, p2n, best);
  nnclr_logits<<<dim3(32, 4, 2), 256, 0, stream>>>(queue, p1n, p2n, best, Mbuf, temp);
  nnclr_loss<<<4 * BB, 256, 0, stream>>>(Mbuf, out);
}
